// SpatialSelfAttention_76227079569554
// MI455X (gfx1250) — matrix operand order repaired and checked by reading, not compiled, not run on hardware
//
#include <hip/hip_runtime.h>
#include <cstdint>

#define C_ 512
#define B_ 8
#define N_ 4096
#define GROUPS_ 32
#define CPG_ 16

typedef __attribute__((ext_vector_type(16))) _Float16 v16h;
typedef __attribute__((ext_vector_type(8)))  float    v8f;

union HTile { uint4 q[2]; v16h h; };
union Acc8  { v8f v; float f[8]; };

__device__ inline v8f wmma_f16(v16h a, v16h b, v8f c) {
  return __builtin_amdgcn_wmma_f32_16x16x32_f16(false, a, false, b, (short)0, c,
                                                false, false);
}

// A tile 16x32 (MxK) from row-major [M][lda] f16.
// ISA layout: lane<16 -> row=lane, K 0-7 & 16-23; lane>=16 -> K 8-15 & 24-31.
__device__ inline v16h load_a(const _Float16* a, int lda) {
  int lane = threadIdx.x & 31;
  int row  = lane & 15;
  int hi   = lane >> 4;
  const _Float16* p = a + (size_t)row * lda + hi * 8;
  HTile t;
  t.q[0] = *(const uint4*)(p);
  t.q[1] = *(const uint4*)(p + 16);
  return t.h;
}

// B tile 32x16 (KxN) from memory laid out [Ncol][ldb] (i.e. B^T row-major).
// ISA layout: lane<16 -> col=lane, K 0-7 & 16-23; lane>=16 -> K 8-15 & 24-31.
__device__ inline v16h load_bt(const _Float16* bt, int ldb) {
  int lane = threadIdx.x & 31;
  int col  = lane & 15;
  int hi   = lane >> 4;
  const _Float16* p = bt + (size_t)col * ldb + hi * 8;
  HTile t;
  t.q[0] = *(const uint4*)(p);
  t.q[1] = *(const uint4*)(p + 16);
  return t.h;
}

// Async global->LDS copy of one 16B chunk per lane (GLOBAL_LOAD_ASYNC_TO_LDS_B128,
// tracked by ASYNCcnt).  LDS byte address = low 32 bits of the generic pointer
// (flat->LDS aperture mapping truncates, ISA 10.2).
__device__ inline void async_copy16(const _Float16* g, _Float16* l) {
  uint32_t laddr = (uint32_t)(size_t)l;
  asm volatile("global_load_async_to_lds_b128 %0, %1, off"
               :: "v"(laddr), "v"(g) : "memory");
}
__device__ inline void async_wait0() {
  asm volatile("s_wait_asynccnt 0x0" ::: "memory");
}

// ---------------- kernel 0: weights f32 -> f16 ----------------
__global__ __launch_bounds__(256) void cvt_weights(const float* wq, const float* wk,
                                                   const float* wv, const float* wo,
                                                   _Float16* w16) {
  const int per = C_ * C_;
  const int total = 4 * per;
  for (int i = blockIdx.x * blockDim.x + threadIdx.x; i < total;
       i += gridDim.x * blockDim.x) {
    int a = i / per, r = i % per;
    const float* src = (a == 0) ? wq : (a == 1) ? wk : (a == 2) ? wv : wo;
    w16[i] = (_Float16)src[r];
  }
}

// ---------------- kernel 1: GroupNorm -> hnT[b][n][c] f16 ----------------
__global__ __launch_bounds__(256) void groupnorm_kernel(const float* x,
                                                        const float* gscale,
                                                        const float* gbias,
                                                        _Float16* hnT) {
  int b = blockIdx.x / GROUPS_;
  int g = blockIdx.x % GROUPS_;
  int tid = threadIdx.x;
  const float* xg = x + ((size_t)b * C_ + g * CPG_) * N_;
  const int total = CPG_ * N_;  // 65536
  float s = 0.f, s2 = 0.f;
  for (int i = tid; i < total; i += 256) {
    float v = xg[i];
    s += v; s2 += v * v;
  }
  __shared__ float rs[256], rs2[256];
  rs[tid] = s; rs2[tid] = s2;
  __syncthreads();
  for (int off = 128; off > 0; off >>= 1) {
    if (tid < off) { rs[tid] += rs[tid + off]; rs2[tid] += rs2[tid + off]; }
    __syncthreads();
  }
  float mean = rs[0] * (1.0f / total);
  float var  = rs2[0] * (1.0f / total) - mean * mean;
  float rstd = rsqrtf(var + 1e-6f);
  for (int i = tid; i < total; i += 256) {
    int cl = i >> 12;          // i / N_
    int n  = i & (N_ - 1);
    int c  = g * CPG_ + cl;
    float y = (xg[i] - mean) * rstd * gscale[c] + gbias[c];
    hnT[((size_t)b * N_ + n) * C_ + c] = (_Float16)y;
  }
}

// ---------------- kernel 2: QKV projections ----------------
// q,k stored transposed [b][n][c]; v stored [b][c][n]. One wave = 16x64 tile.
__global__ __launch_bounds__(256) void qkv_kernel(const _Float16* w16,
                                                  const _Float16* hnT,
                                                  const float* bq, const float* bk,
                                                  const float* bv,
                                                  _Float16* qT, _Float16* kT,
                                                  _Float16* v) {
  int wid = blockIdx.x * 8 + (threadIdx.x >> 5);
  int tn  = wid & 63;
  int tm  = (wid >> 6) & 31;
  int mb  = wid >> 11;          // 0..23
  int mat = mb % 3;
  int b   = mb / 3;
  int lane = threadIdx.x & 31;
  int hi = lane >> 4;

  const _Float16* W = w16 + (size_t)mat * C_ * C_;
  const float* bias = (mat == 0) ? bq : (mat == 1) ? bk : bv;
  int m0 = tm * 16;
  int n0 = tn * 64;

  Acc8 acc[4];
  for (int t = 0; t < 4; ++t)
    for (int j = 0; j < 8; ++j) acc[t].f[j] = 0.0f;

  const _Float16* abase = W + (size_t)m0 * C_;
  const _Float16* bbase = hnT + ((size_t)b * N_ + n0) * C_;

  // fully-unrolled k-loop, parity double-buffered prefetch (no register copies)
  v16h a[2];
  v16h bm[2][4];
  a[0] = load_a(abase, C_);
#pragma unroll
  for (int t = 0; t < 4; ++t) bm[0][t] = load_bt(bbase + (size_t)t * 16 * C_, C_);
#pragma unroll
  for (int kk = 0; kk < C_ / 32; ++kk) {
    const int cur = kk & 1, nxt = cur ^ 1;
    if (kk + 1 < C_ / 32) {
      a[nxt] = load_a(abase + (kk + 1) * 32, C_);
#pragma unroll
      for (int t = 0; t < 4; ++t)
        bm[nxt][t] = load_bt(bbase + (size_t)t * 16 * C_ + (kk + 1) * 32, C_);
    }
#pragma unroll
    for (int t = 0; t < 4; ++t) acc[t].v = wmma_f16(a[cur], bm[cur][t], acc[t].v);
  }

  float bb[8];
  for (int j = 0; j < 8; ++j) bb[j] = bias[m0 + hi * 8 + j];

  if (mat < 2) {
    _Float16* dst = (mat == 0) ? qT : kT;
    for (int t = 0; t < 4; ++t) {
      int n = n0 + t * 16 + (lane & 15);
      union { uint4 q; _Float16 h[8]; } pk;
      for (int j = 0; j < 8; ++j) pk.h[j] = (_Float16)(acc[t].f[j] + bb[j]);
      *(uint4*)(dst + ((size_t)b * N_ + n) * C_ + m0 + hi * 8) = pk.q;
    }
  } else {
    for (int t = 0; t < 4; ++t) {
      int n = n0 + t * 16 + (lane & 15);
      for (int j = 0; j < 8; ++j) {
        int o = m0 + hi * 8 + j;
        v[((size_t)b * C_ + o) * N_ + n] = (_Float16)(acc[t].f[j] + bb[j]);
      }
    }
  }
}

// ---------------- kernel 3: flash attention ----------------
// 1 block (8 waves) per 16-query block. Wave w owns channel slice [w*64, w*64+64).
__global__ __launch_bounds__(256) void attn_kernel(const _Float16* qT,
                                                   const _Float16* kT,
                                                   const _Float16* v,
                                                   _Float16* attnT) {
  __shared__ alignas(16) _Float16 Qs[16 * C_];   // 16 KB, [q][c]
  __shared__ float    Ss[16 * 128];              // 8 KB, scores [q][key]
  __shared__ alignas(16) _Float16 Ps[16 * 128];  // 4 KB, probs  [q][key]
  __shared__ float rowmax[16], rowsum[16], rowscale[16];

  int b  = blockIdx.x >> 8;
  int q0 = (blockIdx.x & 255) * 16;
  int tid  = threadIdx.x;
  int wave = tid >> 5;
  int lane = tid & 31;
  int hi   = lane >> 4;
  int cb   = wave * 64;

  {  // stage Q block (16 x 512 f16 = 1024 chunks of 16B) into LDS, async
    const _Float16* src = qT + ((size_t)b * N_ + q0) * C_;
    for (int i = tid; i < 1024; i += 256)
      async_copy16(src + i * 8, Qs + i * 8);
    async_wait0();
  }
  if (tid < 16) { rowmax[tid] = -1e30f; rowsum[tid] = 0.f; }
  __syncthreads();

  // Q is invariant over all key tiles: keep all 16 A-tiles register-resident
  v16h Qa[16];
#pragma unroll
  for (int kk = 0; kk < 16; ++kk) Qa[kk] = load_a(Qs + kk * 32, C_);  // ds_load_b128

  Acc8 accO[4];
  for (int t = 0; t < 4; ++t)
    for (int j = 0; j < 8; ++j) accO[t].f[j] = 0.0f;

  const float sm_scale = 0.044194173824159216f;  // 1/sqrt(512)

  for (int m0 = 0; m0 < N_; m0 += 128) {
    // ---- S strip: this wave computes 16 queries x 16 keys at m0 + wave*16
    Acc8 s;
    for (int j = 0; j < 8; ++j) s.f[j] = 0.0f;
    const _Float16* kbase = kT + ((size_t)b * N_ + m0 + wave * 16) * C_;
    v16h bm[2];
    bm[0] = load_bt(kbase, C_);
#pragma unroll
    for (int kk = 0; kk < 16; ++kk) {
      const int cur = kk & 1;
      if (kk + 1 < 16) bm[cur ^ 1] = load_bt(kbase + (kk + 1) * 32, C_);
      s.v = wmma_f16(Qa[kk], bm[cur], s.v);
    }
    int col = wave * 16 + (lane & 15);
    for (int j = 0; j < 8; ++j)
      Ss[(hi * 8 + j) * 128 + col] = s.f[j] * sm_scale;
    __syncthreads();

    // ---- online softmax: 16-lane group per row (2 rows per wave), all busy
    {
      int row = wave * 2 + hi;
      int sub = lane & 15;
      float omax = rowmax[row];
      float vals[8];
      float tmax = -1e30f;
      for (int i = 0; i < 8; ++i) {
        vals[i] = Ss[row * 128 + sub + 16 * i];
        tmax = fmaxf(tmax, vals[i]);
      }
      for (int m = 8; m >= 1; m >>= 1)           // stays inside 16-lane half
        tmax = fmaxf(tmax, __shfl_xor(tmax, m));
      float nmax = fmaxf(omax, tmax);
      float scl  = __expf(omax - nmax);
      float sum = 0.f;
      for (int i = 0; i < 8; ++i) {
        float p = __expf(vals[i] - nmax);
        Ps[row * 128 + sub + 16 * i] = (_Float16)p;
        sum += p;
      }
      for (int m = 8; m >= 1; m >>= 1)
        sum += __shfl_xor(sum, m);
      if (sub == 0) {
        rowsum[row]   = rowsum[row] * scl + sum;
        rowmax[row]   = nmax;
        rowscale[row] = scl;
      }
    }
    __syncthreads();

    // ---- rescale running O, then O += P @ V^T for this key tile
    float f[8];
    for (int j = 0; j < 8; ++j) f[j] = rowscale[hi * 8 + j];
    for (int t = 0; t < 4; ++t)
      for (int j = 0; j < 8; ++j) accO[t].f[j] *= f[j];

    const _Float16* vbase = v + ((size_t)b * C_ + cb) * N_ + m0;
#pragma unroll
    for (int ks = 0; ks < 4; ++ks) {
      v16h pa = load_a(Ps + ks * 32, 128);       // ds_load_b128
#pragma unroll
      for (int t = 0; t < 4; ++t) {
        v16h bmv = load_bt(vbase + (size_t)t * 16 * N_ + ks * 32, N_);
        accO[t].v = wmma_f16(pa, bmv, accO[t].v);
      }
    }
    // next score phase writes Ss (not Ps) and is followed by a barrier before
    // Ps is rewritten, so two barriers per tile suffice.
  }

  float inv[8];
  for (int j = 0; j < 8; ++j) inv[j] = 1.0f / rowsum[hi * 8 + j];
  for (int t = 0; t < 4; ++t) {
    int c = cb + t * 16 + (lane & 15);
    for (int j = 0; j < 8; ++j) {
      int qrow = hi * 8 + j;
      attnT[((size_t)b * N_ + q0 + qrow) * C_ + c] = (_Float16)(accO[t].f[j] * inv[j]);
    }
  }
}

// ---------------- kernel 4: output projection + bias + residual ----------------
__global__ __launch_bounds__(256) void proj_kernel(const _Float16* wo16,
                                                   const _Float16* attnT,
                                                   const float* bo, const float* x,
                                                   float* out) {
  int wid = blockIdx.x * 8 + (threadIdx.x >> 5);
  int tn = wid & 63;
  int tm = (wid >> 6) & 31;
  int b  = wid >> 11;
  int lane = threadIdx.x & 31;
  int hi = lane >> 4;
  int m0 = tm * 16, n0 = tn * 64;

  Acc8 acc[4];
  for (int t = 0; t < 4; ++t)
    for (int j = 0; j < 8; ++j) acc[t].f[j] = 0.0f;

  const _Float16* abase = wo16 + (size_t)m0 * C_;
  const _Float16* bbase = attnT + ((size_t)b * N_ + n0) * C_;

  v16h a[2];
  v16h bm[2][4];
  a[0] = load_a(abase, C_);
#pragma unroll
  for (int t = 0; t < 4; ++t) bm[0][t] = load_bt(bbase + (size_t)t * 16 * C_, C_);
#pragma unroll
  for (int kk = 0; kk < C_ / 32; ++kk) {
    const int cur = kk & 1, nxt = cur ^ 1;
    if (kk + 1 < C_ / 32) {
      a[nxt] = load_a(abase + (kk + 1) * 32, C_);
#pragma unroll
      for (int t = 0; t < 4; ++t)
        bm[nxt][t] = load_bt(bbase + (size_t)t * 16 * C_ + (kk + 1) * 32, C_);
    }
#pragma unroll
    for (int t = 0; t < 4; ++t) acc[t].v = wmma_f16(a[cur], bm[cur][t], acc[t].v);
  }

  for (int t = 0; t < 4; ++t) {
    int n = n0 + t * 16 + (lane & 15);
    for (int j = 0; j < 8; ++j) {
      int o = m0 + hi * 8 + j;
      size_t idx = ((size_t)b * C_ + o) * N_ + n;
      out[idx] = acc[t].f[j] + bo[o] + x[idx];
    }
  }
}

extern "C" void kernel_launch(void* const* d_in, const int* in_sizes, int n_in,
                              void* d_out, int out_size, void* d_ws, size_t ws_size,
                              hipStream_t stream) {
  const float* x  = (const float*)d_in[0];
  const float* gs = (const float*)d_in[1];
  const float* gb = (const float*)d_in[2];
  const float* wq = (const float*)d_in[3];
  const float* bq = (const float*)d_in[4];
  const float* wk = (const float*)d_in[5];
  const float* bk = (const float*)d_in[6];
  const float* wv = (const float*)d_in[7];
  const float* bv = (const float*)d_in[8];
  const float* wo = (const float*)d_in[9];
  const float* bo = (const float*)d_in[10];
  float* out = (float*)d_out;

  char* ws = (char*)d_ws;
  _Float16* w16 = (_Float16*)ws;                                   // 4*C*C f16
  _Float16* hnT = (_Float16*)(ws + (size_t)4 * C_ * C_ * sizeof(_Float16));
  _Float16* attnT = hnT;                                           // reuse after QKV
  _Float16* qT = hnT + (size_t)B_ * N_ * C_;
  _Float16* kT = qT + (size_t)B_ * N_ * C_;
  _Float16* vv = kT + (size_t)B_ * N_ * C_;

  cvt_weights<<<dim3(512), dim3(256), 0, stream>>>(wq, wk, wv, wo, w16);
  groupnorm_kernel<<<dim3(B_ * GROUPS_), dim3(256), 0, stream>>>(x, gs, gb, hnT);
  qkv_kernel<<<dim3(6144), dim3(256), 0, stream>>>(w16, hnT, bq, bk, bv, qT, kT, vv);
  attn_kernel<<<dim3(B_ * (N_ / 16)), dim3(256), 0, stream>>>(qT, kT, vv, attnT);
  proj_kernel<<<dim3(2048), dim3(256), 0, stream>>>(w16 + (size_t)3 * C_ * C_,
                                                    attnT, bo, x, out);
}
